// GraphConvolution_57389353009503
// MI455X (gfx1250) — matrix operand order repaired and checked by reading, not compiled, not run on hardware
//
#include <hip/hip_runtime.h>
#include <hip/hip_bf16.h>

typedef __attribute__((ext_vector_type(16))) __bf16 bf16x16;
typedef __attribute__((ext_vector_type(8)))  __bf16 bf16x8;
typedef __attribute__((ext_vector_type(4)))  __bf16 bf16x4;
typedef __attribute__((ext_vector_type(8)))  float  v8f;

#define IN_DIM   256
#define OUT_DIM  128
#define LDS_STRIDE 264   // 256 + 8 bf16 pad -> lane n B-frag @ dword offset 4n mod 64: bank-conflict-free
#define WAVES_PER_BLOCK 8

// ---------------------------------------------------------------------------
// Kernel 1: out[i][j] = bias[j]  (float4-vectorized)
// ---------------------------------------------------------------------------
__global__ void gc_init_kernel(const float* __restrict__ bias,
                               float* __restrict__ out, int total4) {
    int i = blockIdx.x * blockDim.x + threadIdx.x;
    if (i < total4) {
        float4 b = ((const float4*)bias)[i & (OUT_DIM / 4 - 1)];
        ((float4*)out)[i] = b;
    }
}

// ---------------------------------------------------------------------------
// Kernel 2: h = x @ W  via v_wmma_f32_16x16x32_bf16, h stored as bf16.
// One wave -> one 16-row x 128-col strip of h. Weight staged in LDS,
// transposed+padded into B-fragment-friendly layout.
// ---------------------------------------------------------------------------
__global__ void gc_gemm_kernel(const float* __restrict__ x,
                               const float* __restrict__ w,
                               __bf16* __restrict__ h, int nStrips) {
    extern __shared__ __bf16 Wt[];   // [OUT_DIM][LDS_STRIDE]

    // Stage W transposed: Wt[n][k] = (bf16) w[k][n]
    for (int i = threadIdx.x; i < OUT_DIM * IN_DIM; i += blockDim.x) {
        int n = i >> 8;          // i / 256
        int k = i & (IN_DIM - 1);
        Wt[n * LDS_STRIDE + k] = (__bf16)w[k * OUT_DIM + n];
    }
    __syncthreads();

    const int wave = threadIdx.x >> 5;
    const int lane = threadIdx.x & 31;
    const int strip = blockIdx.x * WAVES_PER_BLOCK + wave;
    if (strip >= nStrips) return;

    const int m    = lane & 15;   // row (A) / col (B,C,D) within tile
    const int half = lane >> 4;   // lane group 0: K 0..7,16..23 ; group 1: K 8..15,24..31

    const float* __restrict__ arow = x + (size_t)(strip * 16 + m) * IN_DIM;

    v8f acc[8];
    #pragma unroll
    for (int nt = 0; nt < 8; ++nt) acc[nt] = (v8f){0,0,0,0,0,0,0,0};

    #pragma unroll
    for (int k0 = 0; k0 < IN_DIM; k0 += 32) {
        // ---- A fragment (16x32 bf16) per ISA layout ----
        const int c0 = k0 + half * 8;        // K chunk for VGPRs 0-3
        const int c1 = k0 + 16 + half * 8;   // K chunk for VGPRs 4-7
        float4 a0 = *(const float4*)(arow + c0);
        float4 a1 = *(const float4*)(arow + c0 + 4);
        float4 a2 = *(const float4*)(arow + c1);
        float4 a3 = *(const float4*)(arow + c1 + 4);
        bf16x16 A;
        A[0]  = (__bf16)a0.x;  A[1]  = (__bf16)a0.y;
        A[2]  = (__bf16)a0.z;  A[3]  = (__bf16)a0.w;
        A[4]  = (__bf16)a1.x;  A[5]  = (__bf16)a1.y;
        A[6]  = (__bf16)a1.z;  A[7]  = (__bf16)a1.w;
        A[8]  = (__bf16)a2.x;  A[9]  = (__bf16)a2.y;
        A[10] = (__bf16)a2.z;  A[11] = (__bf16)a2.w;
        A[12] = (__bf16)a3.x;  A[13] = (__bf16)a3.y;
        A[14] = (__bf16)a3.z;  A[15] = (__bf16)a3.w;

        const int kb = k0 + half * 8;        // B: lanes 0-15 K 0..7,16..23 ; lanes 16-31 K 8..15,24..31
        #pragma unroll
        for (int nt = 0; nt < 8; ++nt) {
            const __bf16* bp = &Wt[(nt * 16 + m) * LDS_STRIDE + kb];
            bf16x8 blo = *(const bf16x8*)(bp);
            bf16x8 bhi = *(const bf16x8*)(bp + 16);
            bf16x16 B;
            #pragma unroll
            for (int j = 0; j < 8; ++j) { B[j] = blo[j]; B[j + 8] = bhi[j]; }
            acc[nt] = __builtin_amdgcn_wmma_f32_16x16x32_bf16(
                false, A, false, B, (short)0, acc[nt], false, false);
        }
    }

    // ---- Store D: VGPR r -> row (r + 8*half), col = nt*16 + m ----
    #pragma unroll
    for (int nt = 0; nt < 8; ++nt) {
        #pragma unroll
        for (int r = 0; r < 8; ++r) {
            int row = strip * 16 + r + half * 8;
            h[(size_t)row * OUT_DIM + nt * 16 + m] = (__bf16)acc[nt][r];
        }
    }
}

// ---------------------------------------------------------------------------
// Kernel 3: edge scatter.  One wave per edge; lane covers 4 columns.
// out[dst][c] += val * h[src][c]   via hardware global_atomic_add_f32.
// h (25.6 MB) and out (51.2 MB) are L2-resident on MI455X (192 MB L2).
// ---------------------------------------------------------------------------
__global__ void gc_scatter_kernel(const int* __restrict__ esrc,
                                  const int* __restrict__ edst,
                                  const float* __restrict__ evals,
                                  const __bf16* __restrict__ h,
                                  float* __restrict__ out, int E) {
    int gid  = blockIdx.x * blockDim.x + threadIdx.x;
    int e    = gid >> 5;
    int lane = threadIdx.x & 31;
    if (e >= E) return;

    int   s = esrc[e];
    int   d = edst[e];
    float v = evals[e];

    bf16x4 hv = *(const bf16x4*)(h + (size_t)s * OUT_DIM + lane * 4);
    float* op = out + (size_t)d * OUT_DIM + lane * 4;
    unsafeAtomicAdd(op + 0, v * (float)hv[0]);
    unsafeAtomicAdd(op + 1, v * (float)hv[1]);
    unsafeAtomicAdd(op + 2, v * (float)hv[2]);
    unsafeAtomicAdd(op + 3, v * (float)hv[3]);
}

// ---------------------------------------------------------------------------
extern "C" void kernel_launch(void* const* d_in, const int* in_sizes, int n_in,
                              void* d_out, int out_size, void* d_ws, size_t ws_size,
                              hipStream_t stream) {
    const int*   edge_src  = (const int*)d_in[0];
    const int*   edge_dst  = (const int*)d_in[1];
    const float* edge_vals = (const float*)d_in[2];
    const float* x         = (const float*)d_in[3];
    const float* weight    = (const float*)d_in[4];
    const float* bias      = (const float*)d_in[5];
    float*       out       = (float*)d_out;

    const int E       = in_sizes[0];
    const int nNodes  = in_sizes[3] / IN_DIM;
    const int nStrips = (nNodes + 15) / 16;           // 100000/16 = 6250 exact

    __bf16* h = (__bf16*)d_ws;                        // 25.6 MB bf16 workspace

    // 1) out = bias (broadcast)
    {
        int total4 = nNodes * (OUT_DIM / 4);
        int blocks = (total4 + 255) / 256;
        gc_init_kernel<<<blocks, 256, 0, stream>>>(bias, out, total4);
    }
    // 2) h = bf16(x @ W)
    {
        int blocks = (nStrips + WAVES_PER_BLOCK - 1) / WAVES_PER_BLOCK;
        size_t lds = (size_t)OUT_DIM * LDS_STRIDE * sizeof(__bf16);  // 67,584 B
        gc_gemm_kernel<<<blocks, 32 * WAVES_PER_BLOCK, lds, stream>>>(x, weight, h, nStrips);
    }
    // 3) scatter-accumulate over edges
    {
        long long waves  = E;
        long long blocks = (waves * 32 + 255) / 256;
        gc_scatter_kernel<<<(int)blocks, 256, 0, stream>>>(edge_src, edge_dst, edge_vals,
                                                           h, out, E);
    }
}
